// HOIST_without_claim_25142738551205
// MI455X (gfx1250) — compile-verified, matrix operand order checked by reading, not run on hardware
//
#include <hip/hip_runtime.h>
#include <hip/hip_bf16.h>

// ---------------------------------------------------------------------------
// MI455X (gfx1250, wave32) implementation.
//
// Roofline reasoning:
//  * dist (3000x3000): bandwidth-bound (72MB geo read + 36MB dist write).
//    One block per row, row buffered in LDS (12KB), single-pass softmax.
//  * LSTM: 25 GFLOP across a 64-step serial chain, but rows are independent.
//    Each block owns 16 rows; h/c live in LDS for all 64 steps. Whh (f16,
//    128KB) + lin_w (f16, 32KB) staged in the 320KB WGP LDS once — via the
//    Tensor Data Mover (tensor_load_to_lds, TENSORcnt), not VGPR copies.
//    Recurrent matmul (16x128)@(128x512) done with v_wmma_f32_16x16x32_f16
//    (f16 in / f32 acc): 4 K-chunks x 32 N-tiles, 8 waves x 4 tiles.
//  * Output head fused into the LSTM step (saves 1.5GB of hs traffic):
//    h_att = h*(1+rowsum(dist)) -> 16x128x128 WMMA -> leaky -> dot lin2_w.
// ---------------------------------------------------------------------------

typedef __attribute__((ext_vector_type(16))) _Float16 v16h;
typedef __attribute__((ext_vector_type(8)))  _Float16 v8h;
typedef __attribute__((ext_vector_type(8)))  float    v8f;
typedef __attribute__((ext_vector_type(4)))  unsigned int u32x4;
typedef __attribute__((ext_vector_type(4)))  int i32x4;
typedef __attribute__((ext_vector_type(8)))  int i32x8;

#define NN    3000
#define TT    64
#define DDYN  5
#define RDIM  128
#define FOURR 512
#define MTILE 16

__device__ __forceinline__ float sigm(float x) { return 1.0f / (1.0f + __expf(-x)); }
__device__ __forceinline__ float leaky(float x) { return (x >= 0.f) ? x : 0.01f * x; }

// Load a 16x32 f16 A/B fragment (CDNA5 WMMA register layout) from an LDS row
// of 128 contiguous f16 values. lane%16 selects the row/col (caller bakes it
// into `row`), lane/16 selects which K-halves this lane holds.
__device__ __forceinline__ v16h load_frag(const _Float16* row, int kb, int hi) {
    const v8h lo = *(const v8h*)(row + kb * 32 + hi * 8);
    const v8h hv = *(const v8h*)(row + kb * 32 + 16 + hi * 8);
    v16h r;
#pragma unroll
    for (int q = 0; q < 8; ++q) { r[q] = lo[q]; r[q + 8] = hv[q]; }
    return r;
}

// ---------------------------------------------------------------------------
// Tensor Data Mover: 2D f16 tile (dim0 x dim1 elements, contiguous rows)
// global -> LDS. Descriptor bitfields per CDNA5 ISA section 8.3/8.4:
//   group0: count=1, lds_addr, global_addr[56:0], type=2
//   group1: data_size=1 (2B), tensor_dim0/1, tile_dim0/1 == tensor dims,
//           tensor_dim0_stride = dim0 (packed rows)
// Tracked by TENSORcnt; EXEC ignored, so issue from one wave only.
// ---------------------------------------------------------------------------
__device__ __forceinline__ void tdm_load_2d_f16(const void* gptr, unsigned lds_off,
                                                unsigned dim0, unsigned dim1) {
    unsigned long long ga = (unsigned long long)(size_t)gptr;
    u32x4 g0;
    g0[0] = 1u;                                                 // count=1, user mode
    g0[1] = lds_off;                                            // lds_addr (bytes)
    g0[2] = (unsigned)(ga & 0xFFFFFFFFu);                       // global_addr[31:0]
    g0[3] = (unsigned)((ga >> 32) & 0x01FFFFFFu) | (2u << 30);  // [56:32] + type=2
    i32x8 g1;
    g1[0] = (int)(1u << 16);                                    // data_size=1 -> 2 bytes
    g1[1] = (int)((dim0 & 0xFFFFu) << 16);                      // tensor_dim0 @ bit48
    g1[2] = (int)(((dim0 >> 16) & 0xFFFFu) | ((dim1 & 0xFFFFu) << 16)); // dim0 hi, dim1 lo
    g1[3] = (int)(((dim1 >> 16) & 0xFFFFu) | ((dim0 & 0xFFFFu) << 16)); // dim1 hi, tile_dim0
    g1[4] = (int)(dim1 & 0xFFFFu);                              // tile_dim1, tile_dim2=0
    g1[5] = (int)dim0;                                          // tensor_dim0_stride lo32
    g1[6] = 0;                                                  // stride0 hi, stride1 lo
    g1[7] = 0;
    i32x4 z4 = {0, 0, 0, 0};
#if defined(__clang_major__) && (__clang_major__ >= 23)
    i32x8 z8 = {0, 0, 0, 0, 0, 0, 0, 0};
    __builtin_amdgcn_tensor_load_to_lds(g0, g1, z4, z4, z8, 0);
#else
    __builtin_amdgcn_tensor_load_to_lds(g0, g1, z4, z4, 0);
#endif
}

// ---------------------------------------------------------------------------
// Kernel 1: transpose + f32->f16 convert Whh (128x512) and lin_w (128x128)
// so B fragments are contiguous per output column.
// ---------------------------------------------------------------------------
__global__ void convert_kernel(const float* __restrict__ Whh,
                               const float* __restrict__ lin_w,
                               _Float16* __restrict__ whhT,
                               _Float16* __restrict__ linT) {
    int i = blockIdx.x * blockDim.x + threadIdx.x;
    if (i < FOURR * RDIM) {
        int n = i / RDIM, k = i % RDIM;
        whhT[i] = (_Float16)Whh[k * FOURR + n];
    }
    if (i < RDIM * RDIM) {
        int n = i / RDIM, k = i % RDIM;
        linT[i] = (_Float16)lin_w[k * RDIM + n];
    }
}

// ---------------------------------------------------------------------------
// Kernel 2: per-row pairwise score vectors s1/s2 for pop/demo/eco.
// ---------------------------------------------------------------------------
__global__ void pair_kernel(const float* __restrict__ pop,
                            const float* __restrict__ demo,
                            const float* __restrict__ eco,
                            const float* __restrict__ Wp, const float* __restrict__ ap,
                            const float* __restrict__ Wd, const float* __restrict__ ad,
                            const float* __restrict__ We, const float* __restrict__ ae,
                            float* __restrict__ sp1, float* __restrict__ sp2,
                            float* __restrict__ sd1, float* __restrict__ sd2,
                            float* __restrict__ se1, float* __restrict__ se2) {
    int n = blockIdx.x * blockDim.x + threadIdx.x;
    if (n >= NN) return;
    {   // pop: 16
        float h[16];
#pragma unroll
        for (int k = 0; k < 16; ++k) {
            float s = 0.f;
            for (int j = 0; j < 16; ++j) s += pop[n * 16 + j] * Wp[j * 16 + k];
            h[k] = s;
        }
        float s1 = 0.f, s2 = 0.f;
#pragma unroll
        for (int k = 0; k < 16; ++k) { s1 += h[k] * ap[k]; s2 += h[k] * ap[16 + k]; }
        sp1[n] = s1; sp2[n] = s2;
    }
    {   // demo: 32
        float h[32];
        for (int k = 0; k < 32; ++k) {
            float s = 0.f;
            for (int j = 0; j < 32; ++j) s += demo[n * 32 + j] * Wd[j * 32 + k];
            h[k] = s;
        }
        float s1 = 0.f, s2 = 0.f;
        for (int k = 0; k < 32; ++k) { s1 += h[k] * ad[k]; s2 += h[k] * ad[32 + k]; }
        sd1[n] = s1; sd2[n] = s2;
    }
    {   // eco: 16
        float h[16];
#pragma unroll
        for (int k = 0; k < 16; ++k) {
            float s = 0.f;
            for (int j = 0; j < 16; ++j) s += eco[n * 16 + j] * We[j * 16 + k];
            h[k] = s;
        }
        float s1 = 0.f, s2 = 0.f;
#pragma unroll
        for (int k = 0; k < 16; ++k) { s1 += h[k] * ae[k]; s2 += h[k] * ae[16 + k]; }
        se1[n] = s1; se2[n] = s2;
    }
}

// ---------------------------------------------------------------------------
// Kernel 3: dist row softmax (one block per row; row buffered in LDS).
// Also emits rowsum(dist) per row (used by the fused attention scale).
// ---------------------------------------------------------------------------
__global__ __launch_bounds__(256) void dist_kernel(
    const float* __restrict__ geo,
    const float* __restrict__ Wgeo, const float* __restrict__ ageo,
    const float* __restrict__ sp1, const float* __restrict__ sp2,
    const float* __restrict__ sd1, const float* __restrict__ sd2,
    const float* __restrict__ se1, const float* __restrict__ se2,
    float* __restrict__ dist, float* __restrict__ rs) {
    __shared__ float dbuf[NN];
    __shared__ float red[256];
    const int i = blockIdx.x;
    const int tid = threadIdx.x;
    const float a0 = ageo[0], a1 = ageo[1];
    const float v0 = Wgeo[0] * a0 + Wgeo[1] * a1;
    const float v1 = Wgeo[2] * a0 + Wgeo[3] * a1;
    const float p1 = sp1[i], d1 = sd1[i], e1 = se1[i];

    float lmax = -1e30f;
    for (int j = tid; j < NN; j += 256) {
        const float* g = geo + ((size_t)i * NN + j) * 2;
        if (j + 1024 < NN) __builtin_prefetch(g + 2048, 0, 0);  // global_prefetch_b8
        float d = sigm(p1 + sp2[j]) + sigm(d1 + sd2[j]) + sigm(e1 + se2[j]) +
                  sigm(g[0] * v0 + g[1] * v1);
        dbuf[j] = d;
        lmax = fmaxf(lmax, d);
    }
    red[tid] = lmax; __syncthreads();
    for (int off = 128; off; off >>= 1) {
        if (tid < off) red[tid] = fmaxf(red[tid], red[tid + off]);
        __syncthreads();
    }
    const float m = red[0];
    __syncthreads();

    float lsum = 0.f;
    for (int j = tid; j < NN; j += 256) {
        float e = __expf(dbuf[j] - m);
        dbuf[j] = e;
        lsum += e;
    }
    red[tid] = lsum; __syncthreads();
    for (int off = 128; off; off >>= 1) {
        if (tid < off) red[tid] += red[tid + off];
        __syncthreads();
    }
    const float inv = 1.0f / red[0];
    __syncthreads();

    float lsum2 = 0.f;
    for (int j = tid; j < NN; j += 256) {
        float w = dbuf[j] * inv;
        dist[(size_t)i * NN + j] = w;
        lsum2 += w;
    }
    red[tid] = lsum2; __syncthreads();
    for (int off = 128; off; off >>= 1) {
        if (tid < off) red[tid] += red[tid + off];
        __syncthreads();
    }
    if (tid == 0) rs[i] = red[0];
}

// ---------------------------------------------------------------------------
// Kernel 4: total_weights = [cov_w | hos_w], one wave32 per (n,t).
// ---------------------------------------------------------------------------
__global__ __launch_bounds__(256) void tw_kernel(
    const float* __restrict__ dyn,
    const float* __restrict__ cw1, const float* __restrict__ cb1,
    const float* __restrict__ cw2, const float* __restrict__ cb2,
    const float* __restrict__ hw1, const float* __restrict__ hb1,
    const float* __restrict__ hw2, const float* __restrict__ hb2,
    float* __restrict__ tw) {
    const int gw = (blockIdx.x * 256 + threadIdx.x) >> 5;
    const int lane = threadIdx.x & 31;
    if (gw >= NN * TT) return;
    const int n = gw >> 6, t = gw & 63;
    const float* dp = dyn + ((size_t)n * TT + t) * DDYN;
    const float d0 = dp[0], d1 = dp[1], d2 = dp[2], d3 = dp[3], d4 = dp[4];

    float accc = 0.f, a0 = 0.f, a1 = 0.f, a2 = 0.f, a3 = 0.f;
#pragma unroll
    for (int j = lane; j < RDIM; j += 32) {
        float hc = leaky(d0 * cw1[j] + cb1[j]);
        accc += hc * cw2[j];
        float hh = leaky(d1 * hw1[j] + d2 * hw1[RDIM + j] +
                         d3 * hw1[2 * RDIM + j] + d4 * hw1[3 * RDIM + j] + hb1[j]);
        a0 += hh * hw2[j * 4 + 0];
        a1 += hh * hw2[j * 4 + 1];
        a2 += hh * hw2[j * 4 + 2];
        a3 += hh * hw2[j * 4 + 3];
    }
#pragma unroll
    for (int off = 16; off; off >>= 1) {
        accc += __shfl_xor(accc, off, 32);
        a0 += __shfl_xor(a0, off, 32);
        a1 += __shfl_xor(a1, off, 32);
        a2 += __shfl_xor(a2, off, 32);
        a3 += __shfl_xor(a3, off, 32);
    }
    if (lane == 0) {
        float* o = tw + ((size_t)n * TT + t) * DDYN;
        o[0] = sigm(accc + cb2[0]);
        o[1] = sigm(a0 + hb2[0]);
        o[2] = sigm(a1 + hb2[1]);
        o[3] = sigm(a2 + hb2[2]);
        o[4] = sigm(a3 + hb2[3]);
    }
}

// ---------------------------------------------------------------------------
// Kernel 5: fused LSTM + attention-scaled output head.
// 16 rows per block, 256 threads = 8 waves, ~230KB dynamic LDS.
// ---------------------------------------------------------------------------
static constexpr size_t LSTM_SMEM =
    (size_t)(FOURR * RDIM + RDIM * RDIM + MTILE * RDIM + MTILE * RDIM) * 2 + // f16
    (size_t)(MTILE * RDIM + MTILE * FOURR + MTILE * RDIM + DDYN * FOURR +
             FOURR + RDIM + RDIM + MTILE * DDYN + 256 + MTILE) * 4;          // f32

__global__ __launch_bounds__(256, 1) void lstm_fused_kernel(
    const float* __restrict__ dyn, const float* __restrict__ tw,
    const _Float16* __restrict__ gWhhT, const _Float16* __restrict__ gLinT,
    const float* __restrict__ Wih, const float* __restrict__ bl,
    const float* __restrict__ linb, const float* __restrict__ lin2w,
    const float* __restrict__ lin2b, const float* __restrict__ rs,
    float* __restrict__ y, float* __restrict__ hTo, float* __restrict__ cTo) {
    extern __shared__ __align__(16) char smem[];
    _Float16* sWhh = (_Float16*)smem;            // [512][128] f16 (col-major Whh)
    _Float16* sLin = sWhh + FOURR * RDIM;        // [128][128] f16 (col-major lin_w)
    _Float16* sH   = sLin + RDIM * RDIM;         // [16][128]  h (f16)
    _Float16* sHa  = sH + MTILE * RDIM;          // [16][128]  h*(1+rowsum) (f16)
    float* sC    = (float*)(sHa + MTILE * RDIM); // [16][128]  c
    float* sZ    = sC + MTILE * RDIM;            // [16][512]  z = h@Whh
    float* sU    = sZ + MTILE * FOURR;           // [16][128]  u = ha@lin_w
    float* sWihs = sU + MTILE * RDIM;            // [5][512]
    float* sB    = sWihs + DDYN * FOURR;         // [512]
    float* sLb   = sB + FOURR;                   // [128]
    float* sL2   = sLb + RDIM;                   // [128]
    float* sX    = sL2 + RDIM;                   // [16][5]
    float* sPs   = sX + MTILE * DDYN;            // [16][16]
    float* sRs   = sPs + 256;                    // [16]

    const int tid = threadIdx.x;
    const int rowBase = blockIdx.x * MTILE;

    // Stage the two weight matrices with the Tensor Data Mover (one DMA each,
    // issued by wave 0 only; EXEC is ignored by tensor ops).
    if (tid < 32) {
        tdm_load_2d_f16(gWhhT, (unsigned)(size_t)sWhh, RDIM, FOURR);
        tdm_load_2d_f16(gLinT, (unsigned)(size_t)sLin, RDIM, RDIM);
    }
    // Small arrays via normal loads while the TDM streams in the big tiles.
    for (int i = tid; i < MTILE * RDIM; i += 256) { sH[i] = (_Float16)0.f; sC[i] = 0.f; }
    for (int i = tid; i < DDYN * FOURR; i += 256) sWihs[i] = Wih[i];
    for (int i = tid; i < FOURR; i += 256) sB[i] = bl[i];
    if (tid < RDIM) { sLb[tid] = linb[tid]; sL2[tid] = lin2w[tid]; }
    if (tid < MTILE) {
        int n = rowBase + tid;
        sRs[tid] = (n < NN) ? rs[n] : 0.f;
    }
    if (tid < 32) __builtin_amdgcn_s_wait_tensorcnt(0);
    __syncthreads();

    const int lane = tid & 31;
    const int wv = tid >> 5;       // wave 0..7
    const int m = lane & 15;       // row (A) / col (B) index within tile
    const int hi = lane >> 4;      // K-half select
    const float l2b = lin2b[0];

    for (int t = 0; t < TT; ++t) {
        // --- stage x_t = total_weights * dynamic for the 16 rows ---
        if (tid < MTILE * DDYN) {
            int r = tid / DDYN, d = tid % DDYN;
            int n = rowBase + r;
            float xv = 0.f;
            if (n < NN) {
                size_t o = ((size_t)n * TT + t) * DDYN + d;
                xv = tw[o] * dyn[o];
            }
            sX[tid] = xv;
        }
        __syncthreads();

        // --- z = h @ Whh : M=16, K=128, N=512 via WMMA f16 ---
        v16h afr[4];
#pragma unroll
        for (int kb = 0; kb < 4; ++kb) afr[kb] = load_frag(sH + m * RDIM, kb, hi);
#pragma unroll
        for (int nt = 0; nt < 4; ++nt) {
            const int ncol = (wv * 4 + nt) * 16 + m;
            v8f acc = {};
#pragma unroll
            for (int kb = 0; kb < 4; ++kb) {
                v16h bfr = load_frag(sWhh + ncol * RDIM, kb, hi);
                acc = __builtin_amdgcn_wmma_f32_16x16x32_f16(
                    false, afr[kb], false, bfr, (short)0, acc, false, false);
            }
#pragma unroll
            for (int r = 0; r < 8; ++r)
                sZ[(r + 8 * hi) * FOURR + ncol] = acc[r];
        }
        __syncthreads();

        // --- gates + state update (f32), write back h (f16) ---
        const bool last = (t == TT - 1);
#pragma unroll
        for (int e = 0; e < 8; ++e) {
            int id = tid + e * 256;
            int r = id >> 7, u = id & (RDIM - 1);
            int n = rowBase + r;
            if (n < NN) {
                float xw0 = 0.f, xw1 = 0.f, xw2 = 0.f, xw3 = 0.f;
#pragma unroll
                for (int d = 0; d < DDYN; ++d) {
                    float xv = sX[r * DDYN + d];
                    const float* wr = sWihs + d * FOURR;
                    xw0 += xv * wr[u];
                    xw1 += xv * wr[RDIM + u];
                    xw2 += xv * wr[2 * RDIM + u];
                    xw3 += xv * wr[3 * RDIM + u];
                }
                float zi = sZ[r * FOURR + u]            + sB[u]            + xw0;
                float zf = sZ[r * FOURR + RDIM + u]     + sB[RDIM + u]     + xw1;
                float zg = sZ[r * FOURR + 2 * RDIM + u] + sB[2 * RDIM + u] + xw2;
                float zo = sZ[r * FOURR + 3 * RDIM + u] + sB[3 * RDIM + u] + xw3;
                float ig = sigm(zi), fg = sigm(zf), og = sigm(zo);
                float gg = tanhf(zg);
                float cn = fg * sC[r * RDIM + u] + ig * gg;
                float hn = og * tanhf(cn);
                sC[r * RDIM + u] = cn;
                sH[r * RDIM + u] = (_Float16)hn;
                sHa[r * RDIM + u] = (_Float16)(hn * (1.f + sRs[r]));
                if (last) {
                    hTo[(size_t)n * RDIM + u] = hn;
                    cTo[(size_t)n * RDIM + u] = cn;
                }
            }
        }
        __syncthreads();

        // --- u = h_att @ lin_w : M=16, K=128, N=128 (one 16-col tile/wave) ---
        {
            const int ncol = wv * 16 + m;
            v8f acc = {};
#pragma unroll
            for (int kb = 0; kb < 4; ++kb) {
                v16h af = load_frag(sHa + m * RDIM, kb, hi);
                v16h bf = load_frag(sLin + ncol * RDIM, kb, hi);
                acc = __builtin_amdgcn_wmma_f32_16x16x32_f16(
                    false, af, false, bf, (short)0, acc, false, false);
            }
#pragma unroll
            for (int r = 0; r < 8; ++r)
                sU[(r + 8 * hi) * RDIM + ncol] = acc[r];
        }
        __syncthreads();

        // --- y = leaky(u + lin_b) . lin2_w + lin2_b ---
        {
            int r = tid >> 4, cg = tid & 15;
            float p = 0.f;
#pragma unroll
            for (int q = 0; q < 8; ++q) {
                int u = cg * 8 + q;
                p += leaky(sU[r * RDIM + u] + sLb[u]) * sL2[u];
            }
            sPs[tid] = p;
        }
        __syncthreads();
        if (tid < MTILE) {
            float s = 0.f;
#pragma unroll
            for (int q = 0; q < 16; ++q) s += sPs[tid * 16 + q];
            int n = rowBase + tid;
            if (n < NN) y[(size_t)n * TT + t] = s + l2b;
        }
        __syncthreads();
    }
}

// ---------------------------------------------------------------------------
// Host launcher.
// ---------------------------------------------------------------------------
extern "C" void kernel_launch(void* const* d_in, const int* in_sizes, int n_in,
                              void* d_out, int out_size, void* d_ws, size_t ws_size,
                              hipStream_t stream) {
    const float* dyn    = (const float*)d_in[0];
    const float* pop    = (const float*)d_in[1];
    const float* demo   = (const float*)d_in[2];
    const float* eco    = (const float*)d_in[3];
    const float* geo    = (const float*)d_in[4];
    const float* W_pop  = (const float*)d_in[5];
    const float* a_pop  = (const float*)d_in[6];
    const float* W_demo = (const float*)d_in[7];
    const float* a_demo = (const float*)d_in[8];
    const float* W_eco  = (const float*)d_in[9];
    const float* a_eco  = (const float*)d_in[10];
    const float* W_geo  = (const float*)d_in[11];
    const float* a_geo  = (const float*)d_in[12];
    const float* cw_w1  = (const float*)d_in[13];
    const float* cw_b1  = (const float*)d_in[14];
    const float* cw_w2  = (const float*)d_in[15];
    const float* cw_b2  = (const float*)d_in[16];
    const float* hw_w1  = (const float*)d_in[17];
    const float* hw_b1  = (const float*)d_in[18];
    const float* hw_w2  = (const float*)d_in[19];
    const float* hw_b2  = (const float*)d_in[20];
    const float* Wih    = (const float*)d_in[21];
    const float* Whh    = (const float*)d_in[22];
    const float* b_lstm = (const float*)d_in[23];
    const float* lin_w  = (const float*)d_in[24];
    const float* lin_b  = (const float*)d_in[25];
    const float* lin2_w = (const float*)d_in[26];
    const float* lin2_b = (const float*)d_in[27];

    float* out  = (float*)d_out;
    float* y    = out;                                   // 3000*64
    float* dist = out + NN * TT;                         // 3000*3000
    float* twp  = out + NN * TT + (size_t)NN * NN;       // 3000*64*5
    float* hTo  = twp + (size_t)NN * TT * DDYN;          // 3000*128
    float* cTo  = hTo + (size_t)NN * RDIM;               // 3000*128

    float* wsf = (float*)d_ws;
    float* sp1 = wsf + 0 * NN;
    float* sp2 = wsf + 1 * NN;
    float* sd1 = wsf + 2 * NN;
    float* sd2 = wsf + 3 * NN;
    float* se1 = wsf + 4 * NN;
    float* se2 = wsf + 5 * NN;
    float* rsv = wsf + 6 * NN;
    _Float16* whhT = (_Float16*)((char*)d_ws + 7 * NN * sizeof(float)); // 84000B, 16B aligned
    _Float16* linT = whhT + FOURR * RDIM;

    convert_kernel<<<(FOURR * RDIM + 255) / 256, 256, 0, stream>>>(Whh, lin_w, whhT, linT);

    pair_kernel<<<(NN + 255) / 256, 256, 0, stream>>>(
        pop, demo, eco, W_pop, a_pop, W_demo, a_demo, W_eco, a_eco,
        sp1, sp2, sd1, sd2, se1, se2);

    dist_kernel<<<NN, 256, 0, stream>>>(
        geo, W_geo, a_geo, sp1, sp2, sd1, sd2, se1, se2, dist, rsv);

    tw_kernel<<<(NN * TT) / 8, 256, 0, stream>>>(
        dyn, cw_w1, cw_b1, cw_w2, cw_b2, hw_w1, hw_b1, hw_w2, hw_b2, twp);

    (void)hipFuncSetAttribute((const void*)lstm_fused_kernel,
                              hipFuncAttributeMaxDynamicSharedMemorySize,
                              (int)LSTM_SMEM);
    lstm_fused_kernel<<<(NN + MTILE - 1) / MTILE, 256, LSTM_SMEM, stream>>>(
        dyn, twp, whhT, linT, Wih, b_lstm, lin_b, lin2_w, lin2_b, rsv,
        y, hTo, cTo);
}
